// Conv_Embedding_3_39462159515871
// MI455X (gfx1250) — compile-verified, matrix operand order checked by reading, not run on hardware
//
#include <hip/hip_runtime.h>
#include <hip/hip_bf16.h>

#define NN 100000
#define DD 256
#define EPSV 1e-5f

typedef __attribute__((ext_vector_type(16))) __bf16 v16bf;
typedef __attribute__((ext_vector_type(8)))  float  v8f;
typedef unsigned int u32x4 __attribute__((ext_vector_type(4)));
typedef int          i32x4 __attribute__((ext_vector_type(4)));
typedef int          i32x8 __attribute__((ext_vector_type(8)));

#if defined(__has_builtin)
#if __has_builtin(__builtin_amdgcn_tensor_load_to_lds) && __has_builtin(__builtin_amdgcn_s_wait_tensorcnt)
#define HAVE_TDM 1
#endif
#endif

union BF4 { __bf16 h[4]; uint2 u; };

// ---------------- f32 -> bf16 (row-major passthrough) ----------------
__global__ __launch_bounds__(256) void cvt_bf16_kernel(const float* __restrict__ in,
                                                       __bf16* __restrict__ out, int n4) {
  int t = blockIdx.x * 256 + threadIdx.x;
  if (t >= n4) return;
  float4 v = ((const float4*)in)[t];
  BF4 o;
  o.h[0] = (__bf16)v.x; o.h[1] = (__bf16)v.y; o.h[2] = (__bf16)v.z; o.h[3] = (__bf16)v.w;
  ((uint2*)out)[t] = o.u;
}

// ---------------- pack W (256x256 f32, row-major k*256+n) into WMMA-B bf16 fragments
// Layout: [kt(8)][nt(16)][lane(32)][elem(16)]  (one v16bf per lane per (kt,nt) frag)
__global__ __launch_bounds__(256) void pack_w_kernel(const float* __restrict__ W,
                                                     __bf16* __restrict__ Wp) {
  int p = blockIdx.x * 256 + threadIdx.x;   // 65536 total elements
  int i    =  p        & 15;
  int lane = (p >> 4)  & 31;
  int nt   = (p >> 9)  & 15;
  int kt   =  p >> 13;
  int hs = lane >> 4;
  int n  = nt * 16 + (lane & 15);
  int k  = kt * 32 + hs * 8 + (i < 8 ? i : i + 8);
  Wp[p] = (__bf16)W[k * DD + n];
}

// ---------------- GEMM: C[m,n] = sum_k A[m,k]*W[k,n] + bias[n], bf16 WMMA, f32 acc
__global__ __launch_bounds__(256) void gemm_bias_kernel(
    const __bf16* __restrict__ A,    // N x 256 bf16 row-major
    const __bf16* __restrict__ Wp,   // packed B fragments (65536 bf16 = 128 KB)
    const float*  __restrict__ bias, // 256
    float*        __restrict__ C,    // N x 256 f32 row-major
    int mTiles) {
  __shared__ __bf16 lW[65536];       // 128 KB of the WGP's 320 KB LDS

#ifdef HAVE_TDM
  // Stage whole packed W into LDS with one Tensor Data Mover op (wave 0 issues).
  // D# per CDNA5 ISA: 1-D tensor, data_size=8B,
  // tensor_dim0 = tile_dim0 = 16384 8-byte elements (131072 bytes).
  if (threadIdx.x < 32) {
    unsigned long long ga = (unsigned long long)(uintptr_t)Wp;
    __attribute__((address_space(3))) __bf16* lp =
        (__attribute__((address_space(3))) __bf16*)lW;
    unsigned ldsa = (unsigned)(unsigned long long)lp;
    u32x4 g0;
    g0[0] = 1u;                                   // count=1, user descriptor
    g0[1] = ldsa;                                 // lds_addr
    g0[2] = (unsigned)ga;                         // global_addr[31:0]
    g0[3] = (unsigned)((ga >> 32) & 0x01FFFFFFull) | 0x80000000u; // addr[56:32] | type=2
    i32x8 g1 = {};
    g1[0] = (int)(3u << 16);                      // data_size=8B, workgroup_mask=0
    g1[1] = (int)(16384u << 16);                  // tensor_dim0[15:0] -> bits 63:48
    g1[2] = 0;                                    // tensor_dim0 hi / tensor_dim1 lo
    g1[3] = (int)(16384u << 16);                  // tile_dim0 -> bits 127:112
    i32x4 z4 = {};
#if __has_include(<hip/amd_detail/amd_gfx1250_TDM.h>)
    i32x8 z8 = {};
    __builtin_amdgcn_tensor_load_to_lds(g0, g1, z4, z4, z8, 0);   // clang-23 form
#else
    __builtin_amdgcn_tensor_load_to_lds(g0, g1, z4, z4, 0);       // ROCm 7.2 form
#endif
    __builtin_amdgcn_s_wait_tensorcnt(0);
  }
  __syncthreads();
#else
  {
    const float4* s = (const float4*)Wp;
    float4*       d = (float4*)lW;
    for (int i = threadIdx.x; i < 8192; i += 256) d[i] = s[i];
  }
  __syncthreads();
#endif

  const int lane = threadIdx.x & 31;
  const int hs   = lane >> 4;        // half-wave select for K sub-ranges
  const int ml   = lane & 15;
  const v16bf* lv = (const v16bf*)lW;

  int waveId   = blockIdx.x * 8 + (threadIdx.x >> 5);
  int numWaves = gridDim.x * 8;
  int totalUnits = mTiles * 4;       // each unit: 16 rows x 64 cols

  for (int unit = waveId; unit < totalUnits; unit += numWaves) {
    int mt = unit >> 2;
    int ng = unit & 3;
    const __bf16* arow = A + (size_t)(mt * 16 + ml) * DD;

    v8f c0 = {}, c1 = {}, c2 = {}, c3 = {};
#pragma unroll
    for (int kt = 0; kt < 8; ++kt) {
      // A fragment: lane<16 holds K {kt*32+0..7, kt*32+16..23}; lane>=16 the +8 halves
      v16bf a;
      const __bf16* p = arow + kt * 32 + hs * 8;
      *((float4*)&a)     = *(const float4*)p;
      *((float4*)&a + 1) = *(const float4*)(p + 16);

      int base = (kt * 16 + ng * 4) * 32 + lane;
      c0 = __builtin_amdgcn_wmma_f32_16x16x32_bf16(false, a, false, lv[base     ], (short)0, c0, false, false);
      c1 = __builtin_amdgcn_wmma_f32_16x16x32_bf16(false, a, false, lv[base + 32], (short)0, c1, false, false);
      c2 = __builtin_amdgcn_wmma_f32_16x16x32_bf16(false, a, false, lv[base + 64], (short)0, c2, false, false);
      c3 = __builtin_amdgcn_wmma_f32_16x16x32_bf16(false, a, false, lv[base + 96], (short)0, c3, false, false);
    }

    int n0 = ng * 64 + ml;
    float bv0 = bias[n0], bv1 = bias[n0 + 16], bv2 = bias[n0 + 32], bv3 = bias[n0 + 48];
    int rbase = mt * 16 + hs * 8;    // C layout: vgpr r -> row r (lanes 0-15) / r+8 (lanes 16-31)
#pragma unroll
    for (int r = 0; r < 8; ++r) {
      float* crow = C + (size_t)(rbase + r) * DD + n0;
      crow[0]  = c0[r] + bv0;
      crow[16] = c1[r] + bv1;
      crow[32] = c2[r] + bv2;
      crow[48] = c3[r] + bv3;
    }
  }
}

// ---------------- SpMM: acc[rows[e],:] += vals[e] * tmp[cols[e],:]  (one wave / edge)
__global__ __launch_bounds__(256) void spmm_kernel(
    const int* __restrict__ rows, const int* __restrict__ cols,
    const float* __restrict__ vals, const float* __restrict__ tmp,
    float* __restrict__ acc, int nEdges) {
  int wave     = (blockIdx.x * 256 + threadIdx.x) >> 5;
  int lane     = threadIdx.x & 31;
  int numWaves = (gridDim.x * 256) >> 5;
  for (int e = wave; e < nEdges; e += numWaves) {
    int   r = rows[e];
    int   c = cols[e];
    float v = vals[e];
    // software prefetch of the next grid-stride iteration's gathered row
    int en = e + numWaves;
    if (en < nEdges) {
      int cn = cols[en];
      __builtin_prefetch(tmp + (size_t)cn * DD + lane * 8, 0, 0);
    }
    const float4* src = (const float4*)(tmp + (size_t)c * DD);
    float*       drow = acc + (size_t)r * DD;
#pragma unroll
    for (int j = 0; j < 2; ++j) {
      float4 x = src[lane + 32 * j];
      int base = (lane + 32 * j) * 4;
      atomicAdd(drow + base + 0, v * x.x);
      atomicAdd(drow + base + 1, v * x.y);
      atomicAdd(drow + base + 2, v * x.z);
      atomicAdd(drow + base + 3, v * x.w);
    }
  }
}

// ---------------- ReLU + LayerNorm (one wave / row); writes f32 e_k and bf16 next input
__global__ __launch_bounds__(256) void relu_ln_kernel(
    const float* __restrict__ acc, const float* __restrict__ g,
    const float* __restrict__ be, float* __restrict__ e_out,
    __bf16* __restrict__ hnext, int nrows) {
  int wave     = (blockIdx.x * 256 + threadIdx.x) >> 5;
  int lane     = threadIdx.x & 31;
  int numWaves = (gridDim.x * 256) >> 5;
  for (int r = wave; r < nrows; r += numWaves) {
    const float4* src = (const float4*)(acc + (size_t)r * DD);
    float4 h0 = src[lane], h1 = src[lane + 32];
    h0.x = fmaxf(h0.x, 0.f); h0.y = fmaxf(h0.y, 0.f);
    h0.z = fmaxf(h0.z, 0.f); h0.w = fmaxf(h0.w, 0.f);
    h1.x = fmaxf(h1.x, 0.f); h1.y = fmaxf(h1.y, 0.f);
    h1.z = fmaxf(h1.z, 0.f); h1.w = fmaxf(h1.w, 0.f);
    float sum = h0.x + h0.y + h0.z + h0.w + h1.x + h1.y + h1.z + h1.w;
    float sq  = h0.x*h0.x + h0.y*h0.y + h0.z*h0.z + h0.w*h0.w
              + h1.x*h1.x + h1.y*h1.y + h1.z*h1.z + h1.w*h1.w;
#pragma unroll
    for (int off = 16; off > 0; off >>= 1) {
      sum += __shfl_xor(sum, off, 32);
      sq  += __shfl_xor(sq,  off, 32);
    }
    float mu  = sum * (1.0f / 256.0f);
    float inv = rsqrtf(sq * (1.0f / 256.0f) - mu * mu + EPSV);

    float4 G0 = ((const float4*)g)[lane],  G1 = ((const float4*)g)[lane + 32];
    float4 B0 = ((const float4*)be)[lane], B1 = ((const float4*)be)[lane + 32];
    float4 y0, y1;
    y0.x = (h0.x - mu) * inv * G0.x + B0.x;  y0.y = (h0.y - mu) * inv * G0.y + B0.y;
    y0.z = (h0.z - mu) * inv * G0.z + B0.z;  y0.w = (h0.w - mu) * inv * G0.w + B0.w;
    y1.x = (h1.x - mu) * inv * G1.x + B1.x;  y1.y = (h1.y - mu) * inv * G1.y + B1.y;
    y1.z = (h1.z - mu) * inv * G1.z + B1.z;  y1.w = (h1.w - mu) * inv * G1.w + B1.w;

    float4* eo = (float4*)(e_out + (size_t)r * DD);
    eo[lane]      = y0;
    eo[lane + 32] = y1;

    BF4 o0, o1;
    o0.h[0] = (__bf16)y0.x; o0.h[1] = (__bf16)y0.y; o0.h[2] = (__bf16)y0.z; o0.h[3] = (__bf16)y0.w;
    o1.h[0] = (__bf16)y1.x; o1.h[1] = (__bf16)y1.y; o1.h[2] = (__bf16)y1.z; o1.h[3] = (__bf16)y1.w;
    uint2* ho = (uint2*)(hnext + (size_t)r * DD);
    ho[lane]      = o0.u;
    ho[lane + 32] = o1.u;
  }
}

// ---------------- gather rows of [e1|e2|e3] + recon_loss = 0
__global__ __launch_bounds__(256) void gather_kernel(
    const int* __restrict__ x, const float* __restrict__ e1,
    const float* __restrict__ e2, const float* __restrict__ e3,
    float* __restrict__ out, int Bq) {
  int t = blockIdx.x * 256 + threadIdx.x;
  if (t == 0) out[(size_t)Bq * 768] = 0.0f;   // recon_loss
  int total = Bq * 192;                       // 192 float4s per output row
  if (t >= total) return;
  int b = t / 192;
  int j = t - b * 192;
  int idx = x[b] - 1;
  const float* src; int jj;
  if (j < 64)       { src = e1; jj = j; }
  else if (j < 128) { src = e2; jj = j - 64; }
  else              { src = e3; jj = j - 128; }
  float4 v = ((const float4*)(src + (size_t)idx * DD))[jj];
  ((float4*)(out + (size_t)b * 768))[j] = v;
}

extern "C" void kernel_launch(void* const* d_in, const int* in_sizes, int n_in,
                              void* d_out, int out_size, void* d_ws, size_t ws_size,
                              hipStream_t stream) {
  const int*   x    = (const int*)d_in[0];
  const int*   rows = (const int*)d_in[1];
  const int*   cols = (const int*)d_in[2];
  const float* vals = (const float*)d_in[3];
  const float* emb  = (const float*)d_in[4];
  const float* Wt[3] = {(const float*)d_in[5], (const float*)d_in[9],  (const float*)d_in[13]};
  const float* bt[3] = {(const float*)d_in[6], (const float*)d_in[10], (const float*)d_in[14]};
  const float* gt[3] = {(const float*)d_in[7], (const float*)d_in[11], (const float*)d_in[15]};
  const float* bet[3]= {(const float*)d_in[8], (const float*)d_in[12], (const float*)d_in[16]};
  int nEdges = in_sizes[1];
  int Bq     = in_sizes[0];

  char*  ws     = (char*)d_ws;
  size_t eBytes = (size_t)NN * DD * 4;              // 102.4 MB
  float*  e1  = (float*)(ws);
  float*  e2  = (float*)(ws + eBytes);
  float*  e3  = (float*)(ws + 2 * eBytes);
  float*  tmp = (float*)(ws + 3 * eBytes);
  float*  acc = (float*)(ws + 4 * eBytes);
  __bf16* hA  = (__bf16*)(ws + 5 * eBytes);          // N x 256 bf16
  __bf16* Wp  = (__bf16*)(ws + 5 * eBytes + (size_t)NN * DD * 2);
  float*  eouts[3] = {e1, e2, e3};

  // layer-1 input: embeddings -> bf16
  int n4 = NN * DD / 4;
  cvt_bf16_kernel<<<(n4 + 255) / 256, 256, 0, stream>>>(emb, hA, n4);

  for (int L = 0; L < 3; ++L) {
    pack_w_kernel<<<256, 256, 0, stream>>>(Wt[L], Wp);
    gemm_bias_kernel<<<512, 256, 0, stream>>>(hA, Wp, bt[L], tmp, NN / 16);
    hipMemsetAsync(acc, 0, eBytes, stream);
    spmm_kernel<<<8192, 256, 0, stream>>>(rows, cols, vals, tmp, acc, nEdges);
    relu_ln_kernel<<<4096, 256, 0, stream>>>(acc, gt[L], bet[L], eouts[L], hA, NN);
  }

  gather_kernel<<<(Bq * 192 + 255) / 256, 256, 0, stream>>>(x, e1, e2, e3, (float*)d_out, Bq);
}
